// CrossAttention_15590731284898
// MI455X (gfx1250) — matrix operand order repaired and checked by reading, not compiled, not run on hardware
//
#include <hip/hip_runtime.h>

#define DIMD 1024
#define NHEADS 16
#define HDIM 64
#define NQ 2048
#define NCTX 2048
#define NBATCH 2

typedef __attribute__((ext_vector_type(16))) _Float16 v16h;
typedef __attribute__((ext_vector_type(8)))  _Float16 v8h;
typedef __attribute__((ext_vector_type(8)))  float    v8f;

// ---- WMMA fragment helpers (gfx1250 wave32 layouts, CDNA5 ISA) ----

// A-matrix 16x32 f16 (this code's contiguous k order, not ISA table): lane L<16 holds M=L, halves0..15=K0..15;
// lane L>=16 holds M=L-16, halves0..15=K16..31.
static __device__ __forceinline__ v16h load_a_frag(const _Float16* base, int ld) {
  int lane = threadIdx.x & 31;
  int m  = lane & 15;
  int kb = (lane & 16) ? 16 : 0;
  const _Float16* p = base + (size_t)m * ld + kb;
  return *(const v16h*)p;           // K = kb .. kb+15
}

// B-matrix 32x16 f16 (from transposed [N][K] storage): lane = N column,
// lanes 0-15 hold K=0..15, lanes 16-31 hold K=16..31 (16 contiguous halves; this code's order, as in A).
static __device__ __forceinline__ v16h load_b_frag(const _Float16* baseT, int ld) {
  int lane = threadIdx.x & 31;
  int n  = lane & 15;
  int kb = (lane & 16) ? 16 : 0;
  return *(const v16h*)(baseT + (size_t)n * ld + kb);
}

static __device__ __forceinline__ v8f wmma16(v16h a, v16h b, v8f c) {
  return __builtin_amdgcn_wmma_f32_16x16x32_f16(false, a, false, b, (short)0, c, false, false);
}

// ---- Kernel 1: LayerNorm (f32 in) -> f16 out -------------------------------
__global__ void ln_to_h(const float* __restrict__ x, const float* __restrict__ g,
                        const float* __restrict__ be, _Float16* __restrict__ out) {
  __shared__ float redS[8], redQ[8];
  int row = blockIdx.x;
  int t = threadIdx.x;
  const float* xr = x + (size_t)row * DIMD;
  float v[4];
  float s = 0.f, ss = 0.f;
#pragma unroll
  for (int i = 0; i < 4; ++i) {
    v[i] = xr[t + i * 256];
    s += v[i]; ss += v[i] * v[i];
  }
  for (int m = 16; m >= 1; m >>= 1) {
    s  += __shfl_xor(s, m, 32);
    ss += __shfl_xor(ss, m, 32);
  }
  int wid = t >> 5;
  if ((t & 31) == 0) { redS[wid] = s; redQ[wid] = ss; }
  __syncthreads();
  float st = 0.f, sq = 0.f;
#pragma unroll
  for (int i = 0; i < 8; ++i) { st += redS[i]; sq += redQ[i]; }
  float mu   = st * (1.0f / DIMD);
  float var  = sq * (1.0f / DIMD) - mu * mu;
  float rstd = rsqrtf(var + 1e-5f);
  _Float16* o = out + (size_t)row * DIMD;
#pragma unroll
  for (int i = 0; i < 4; ++i) {
    int c = t + i * 256;
    o[c] = (_Float16)((v[i] - mu) * rstd * g[c] + be[c]);
  }
}

// ---- Kernel 2: weight f32 [K][N] -> f16 transposed [N][K] ------------------
__global__ void wt_convert(const float* __restrict__ W, _Float16* __restrict__ Wt) {
  int idx = blockIdx.x * 256 + threadIdx.x;   // over D*D
  int k = idx >> 10;
  int n = idx & (DIMD - 1);
  Wt[(size_t)n * DIMD + k] = (_Float16)W[idx];
}

// ---- Kernel 3: projection GEMM, 32x64 tile per wave, ping-pong buffered ----
// mode 0: out[b][h][n][d]  (Q, K layouts)   mode 1: out[b][h][d][n] (V^T)
__global__ __launch_bounds__(256, 1)
void proj_gemm(const _Float16* __restrict__ A, const _Float16* __restrict__ Bt,
               const float* __restrict__ bias, _Float16* __restrict__ out,
               int mode) {
  int wid = threadIdx.x >> 5;
  int m0 = (blockIdx.x * 8 + wid) * 32;
  int n0 = blockIdx.y * 64;
  v8f z = {0.f,0.f,0.f,0.f,0.f,0.f,0.f,0.f};
  v8f acc[2][4] = {{z, z, z, z}, {z, z, z, z}};
  const _Float16* A0 = A + (size_t)m0 * DIMD;
  const _Float16* A1 = A0 + 16 * DIMD;
  const _Float16* B0 = Bt + (size_t)n0 * DIMD;

  // buffer set X: fragments for kk
  v16h a0X = load_a_frag(A0, DIMD);
  v16h a1X = load_a_frag(A1, DIMD);
  v16h bX[4];
#pragma unroll
  for (int j = 0; j < 4; ++j) bX[j] = load_b_frag(B0 + (size_t)(16 * j) * DIMD, DIMD);

  for (int kk = 0; kk < DIMD; kk += 64) {
    // buffer set Y: fragments for kk+32 (loads overlap X's WMMAs)
    v16h a0Y = load_a_frag(A0 + kk + 32, DIMD);
    v16h a1Y = load_a_frag(A1 + kk + 32, DIMD);
    v16h bY[4];
#pragma unroll
    for (int j = 0; j < 4; ++j)
      bY[j] = load_b_frag(B0 + (size_t)(16 * j) * DIMD + kk + 32, DIMD);

#pragma unroll
    for (int j = 0; j < 4; ++j) {
      acc[0][j] = wmma16(a0X, bX[j], acc[0][j]);
      acc[1][j] = wmma16(a1X, bX[j], acc[1][j]);
    }

    if (kk + 64 < DIMD) {   // buffer set X: fragments for kk+64
      a0X = load_a_frag(A0 + kk + 64, DIMD);
      a1X = load_a_frag(A1 + kk + 64, DIMD);
#pragma unroll
      for (int j = 0; j < 4; ++j)
        bX[j] = load_b_frag(B0 + (size_t)(16 * j) * DIMD + kk + 64, DIMD);
    }

#pragma unroll
    for (int j = 0; j < 4; ++j) {
      acc[0][j] = wmma16(a0Y, bY[j], acc[0][j]);
      acc[1][j] = wmma16(a1Y, bY[j], acc[1][j]);
    }
  }

  int lane = threadIdx.x & 31;
  int nl = lane & 15;
  int r0 = (lane & 16) ? 8 : 0;
#pragma unroll
  for (int t = 0; t < 2; ++t) {
#pragma unroll
    for (int j = 0; j < 4; ++j) {
#pragma unroll
      for (int i = 0; i < 8; ++i) {
        int row = m0 + 16 * t + r0 + i;
        int col = n0 + 16 * j + nl;
        float v = acc[t][j][i] + bias[col];
        int bb = row >> 11;            // Nseq = 2048
        int q  = row & 2047;
        int h  = col >> 6;
        int d  = col & 63;
        size_t idx;
        if (mode == 0) idx = (((size_t)bb * NHEADS + h) * NQ + q) * HDIM + d;
        else           idx = (((size_t)bb * NHEADS + h) * HDIM + d) * NCTX + q;
        out[idx] = (_Float16)v;
      }
    }
  }
}

// ---- Kernel 4: flash attention, one wave per 16-query tile per (b,h) -------
__global__ __launch_bounds__(256, 1)
void attn_kernel(const _Float16* __restrict__ Q, const _Float16* __restrict__ K,
                 const _Float16* __restrict__ Vt, _Float16* __restrict__ O) {
  __shared__ __align__(32) _Float16 pbuf[8 * 16 * 32];
  int wid = threadIdx.x >> 5;
  int w  = blockIdx.x * 8 + wid;
  int qt = w & 127;     // NQ/16 = 128 tiles
  int bh = w >> 7;      // 0..31
  const _Float16* Qbh = Q  + (size_t)bh * NQ * HDIM;
  const _Float16* Kbh = K  + (size_t)bh * NCTX * HDIM;
  const _Float16* Vbh = Vt + (size_t)bh * HDIM * NCTX;
  _Float16* pb = pbuf + wid * 512;

  v16h aq0 = load_a_frag(Qbh + (size_t)qt * 16 * HDIM + 0,  HDIM);
  v16h aq1 = load_a_frag(Qbh + (size_t)qt * 16 * HDIM + 32, HDIM);

  v8f z = {0.f,0.f,0.f,0.f,0.f,0.f,0.f,0.f};
  v8f oacc[4] = {z, z, z, z};
  float ms[8], ls[8];
#pragma unroll
  for (int i = 0; i < 8; ++i) { ms[i] = -1e30f; ls[i] = 0.f; }

  int lane = threadIdx.x & 31;
  int nl = lane & 15;
  int r0 = (lane & 16) ? 8 : 0;
  const float sc = 0.125f;   // 1/sqrt(64)

  // prologue: K fragments for c0 = 0
  v16h kb00 = load_b_frag(Kbh + 0, HDIM);
  v16h kb01 = load_b_frag(Kbh + 32, HDIM);
  v16h kb10 = load_b_frag(Kbh + (size_t)16 * HDIM + 0,  HDIM);
  v16h kb11 = load_b_frag(Kbh + (size_t)16 * HDIM + 32, HDIM);

  for (int c0 = 0; c0 < NCTX; c0 += 32) {
    // S tiles: 16 queries x 32 context columns
    v8f s0 = z, s1 = z;
    s0 = wmma16(aq0, kb00, s0);
    s0 = wmma16(aq1, kb01, s0);
    s1 = wmma16(aq0, kb10, s1);
    s1 = wmma16(aq1, kb11, s1);

    // V fragments for this step: independent of softmax -> issue now
    v16h bv[4];
#pragma unroll
    for (int j = 0; j < 4; ++j)
      bv[j] = load_b_frag(Vbh + (size_t)(16 * j) * NCTX + c0, NCTX);

    // prefetch next step's K fragments; their latency hides under softmax VALU
    if (c0 + 32 < NCTX) {
      const _Float16* Kn = Kbh + (size_t)(c0 + 32) * HDIM;
      kb00 = load_b_frag(Kn + 0, HDIM);
      kb01 = load_b_frag(Kn + 32, HDIM);
      kb10 = load_b_frag(Kn + 16 * HDIM + 0,  HDIM);
      kb11 = load_b_frag(Kn + 16 * HDIM + 32, HDIM);
    }

    float p0[8], p1[8], so[8];
#pragma unroll
    for (int i = 0; i < 8; ++i) {
      float x0 = s0[i] * sc, x1 = s1[i] * sc;
      float tm = fmaxf(x0, x1);
      tm = fmaxf(tm, __shfl_xor(tm, 1, 32));
      tm = fmaxf(tm, __shfl_xor(tm, 2, 32));
      tm = fmaxf(tm, __shfl_xor(tm, 4, 32));
      tm = fmaxf(tm, __shfl_xor(tm, 8, 32));
      float mnew = fmaxf(ms[i], tm);
      so[i] = __expf(ms[i] - mnew);
      p0[i] = __expf(x0 - mnew);
      p1[i] = __expf(x1 - mnew);
      float ts = p0[i] + p1[i];
      ts += __shfl_xor(ts, 1, 32);
      ts += __shfl_xor(ts, 2, 32);
      ts += __shfl_xor(ts, 4, 32);
      ts += __shfl_xor(ts, 8, 32);
      ls[i] = ls[i] * so[i] + ts;
      ms[i] = mnew;
    }
    // C-layout -> A-layout via LDS bounce; rescale running O accumulators
#pragma unroll
    for (int i = 0; i < 8; ++i) {
      pb[(r0 + i) * 32 + nl]      = (_Float16)p0[i];
      pb[(r0 + i) * 32 + 16 + nl] = (_Float16)p1[i];
      oacc[0][i] *= so[i]; oacc[1][i] *= so[i];
      oacc[2][i] *= so[i]; oacc[3][i] *= so[i];
    }
    asm volatile("s_wait_dscnt 0" ::: "memory");
    v16h ap = load_a_frag(pb, 32);
#pragma unroll
    for (int j = 0; j < 4; ++j)
      oacc[j] = wmma16(ap, bv[j], oacc[j]);
  }

  int b = bh >> 4, h = bh & 15;
#pragma unroll
  for (int j = 0; j < 4; ++j) {
#pragma unroll
    for (int i = 0; i < 8; ++i) {
      int q   = qt * 16 + r0 + i;
      int col = h * 64 + 16 * j + nl;
      O[((size_t)(b * NQ + q)) * DIMD + col] = (_Float16)(oacc[j][i] / ls[i]);
    }
  }
}

// ---- Kernel 5: output GEMM + bias + residual, f32 store --------------------
__global__ __launch_bounds__(256, 1)
void out_gemm(const _Float16* __restrict__ A, const _Float16* __restrict__ Bt,
              const float* __restrict__ bias, const float* __restrict__ resid,
              float* __restrict__ out) {
  int wid = threadIdx.x >> 5;
  int m0 = (blockIdx.x * 8 + wid) * 32;
  int n0 = blockIdx.y * 64;
  v8f z = {0.f,0.f,0.f,0.f,0.f,0.f,0.f,0.f};
  v8f acc[2][4] = {{z, z, z, z}, {z, z, z, z}};
  const _Float16* A0 = A + (size_t)m0 * DIMD;
  const _Float16* A1 = A0 + 16 * DIMD;
  const _Float16* B0 = Bt + (size_t)n0 * DIMD;

  v16h a0X = load_a_frag(A0, DIMD);
  v16h a1X = load_a_frag(A1, DIMD);
  v16h bX[4];
#pragma unroll
  for (int j = 0; j < 4; ++j) bX[j] = load_b_frag(B0 + (size_t)(16 * j) * DIMD, DIMD);

  for (int kk = 0; kk < DIMD; kk += 64) {
    v16h a0Y = load_a_frag(A0 + kk + 32, DIMD);
    v16h a1Y = load_a_frag(A1 + kk + 32, DIMD);
    v16h bY[4];
#pragma unroll
    for (int j = 0; j < 4; ++j)
      bY[j] = load_b_frag(B0 + (size_t)(16 * j) * DIMD + kk + 32, DIMD);

#pragma unroll
    for (int j = 0; j < 4; ++j) {
      acc[0][j] = wmma16(a0X, bX[j], acc[0][j]);
      acc[1][j] = wmma16(a1X, bX[j], acc[1][j]);
    }

    if (kk + 64 < DIMD) {
      a0X = load_a_frag(A0 + kk + 64, DIMD);
      a1X = load_a_frag(A1 + kk + 64, DIMD);
#pragma unroll
      for (int j = 0; j < 4; ++j)
        bX[j] = load_b_frag(B0 + (size_t)(16 * j) * DIMD + kk + 64, DIMD);
    }

#pragma unroll
    for (int j = 0; j < 4; ++j) {
      acc[0][j] = wmma16(a0Y, bY[j], acc[0][j]);
      acc[1][j] = wmma16(a1Y, bY[j], acc[1][j]);
    }
  }

  int lane = threadIdx.x & 31;
  int nl = lane & 15;
  int r0 = (lane & 16) ? 8 : 0;
#pragma unroll
  for (int t = 0; t < 2; ++t) {
#pragma unroll
    for (int j = 0; j < 4; ++j) {
#pragma unroll
      for (int i = 0; i < 8; ++i) {
        int row = m0 + 16 * t + r0 + i;
        int col = n0 + 16 * j + nl;
        size_t idx = (size_t)row * DIMD + col;
        out[idx] = acc[t][j][i] + bias[col] + resid[idx];
      }
    }
  }
}

// ---- Launch ----------------------------------------------------------------
extern "C" void kernel_launch(void* const* d_in, const int* in_sizes, int n_in,
                              void* d_out, int out_size, void* d_ws, size_t ws_size,
                              hipStream_t stream) {
  (void)in_sizes; (void)n_in; (void)out_size; (void)ws_size;
  const float* query   = (const float*)d_in[0];
  const float* context = (const float*)d_in[1];
  const float* Wq = (const float*)d_in[2];
  const float* bq = (const float*)d_in[3];
  const float* Wk = (const float*)d_in[4];
  const float* bk = (const float*)d_in[5];
  const float* Wv = (const float*)d_in[6];
  const float* bv = (const float*)d_in[7];
  const float* Wo = (const float*)d_in[8];
  const float* bo = (const float*)d_in[9];
  const float* gq     = (const float*)d_in[10];
  const float* betaq  = (const float*)d_in[11];
  const float* gkv    = (const float*)d_in[12];
  const float* betakv = (const float*)d_in[13];
  float* out = (float*)d_out;

  char* ws = (char*)d_ws;
  size_t off = 0;
  auto alloc = [&](size_t bytes) {
    char* p = ws + off;
    off += (bytes + 255) & ~(size_t)255;
    return p;
  };
  const size_t actB = (size_t)NBATCH * NQ * DIMD * sizeof(_Float16);   // 8 MiB
  const size_t wB   = (size_t)DIMD * DIMD * sizeof(_Float16);          // 2 MiB
  _Float16* lnq  = (_Float16*)alloc(actB);
  _Float16* lnc  = (_Float16*)alloc(actB);
  _Float16* WqT  = (_Float16*)alloc(wB);
  _Float16* WkT  = (_Float16*)alloc(wB);
  _Float16* WvT  = (_Float16*)alloc(wB);
  _Float16* WoT  = (_Float16*)alloc(wB);
  _Float16* Qh   = (_Float16*)alloc(actB);
  _Float16* Kh   = (_Float16*)alloc(actB);
  _Float16* Vth  = (_Float16*)alloc(actB);
  _Float16* Ah   = (_Float16*)alloc(actB);

  ln_to_h<<<NBATCH * NQ,   256, 0, stream>>>(query,   gq,  betaq,  lnq);
  ln_to_h<<<NBATCH * NCTX, 256, 0, stream>>>(context, gkv, betakv, lnc);

  int cvtBlocks = (DIMD * DIMD) / 256;
  wt_convert<<<cvtBlocks, 256, 0, stream>>>(Wq, WqT);
  wt_convert<<<cvtBlocks, 256, 0, stream>>>(Wk, WkT);
  wt_convert<<<cvtBlocks, 256, 0, stream>>>(Wv, WvT);
  wt_convert<<<cvtBlocks, 256, 0, stream>>>(Wo, WoT);

  dim3 gg(NBATCH * NQ / 32 / 8, DIMD / 64);   // (16, 16)
  proj_gemm<<<gg, 256, 0, stream>>>(lnq, WqT, bq, Qh,  0);
  proj_gemm<<<gg, 256, 0, stream>>>(lnc, WkT, bk, Kh,  0);
  proj_gemm<<<gg, 256, 0, stream>>>(lnc, WvT, bv, Vth, 1);

  attn_kernel<<<NBATCH * NHEADS * (NQ / 16) / 8, 256, 0, stream>>>(Qh, Kh, Vth, Ah);

  out_gemm<<<gg, 256, 0, stream>>>(Ah, WoT, bo, query, out);
}
